// TOICNN_6674379178728
// MI455X (gfx1250) — compile-verified, matrix operand order checked by reading, not run on hardware
//
#include <hip/hip_runtime.h>

// ---------------------------------------------------------------------------
// TOI-CNN forward for MI455X (gfx1250, wave32, WMMA).
// GEMMs: v_wmma_f32_16x16x32_bf16, 128x128 block tile, BK=64, double-buffered
// LDS filled by async global->LDS loads (ASYNCcnt) for latency hiding.
// ---------------------------------------------------------------------------

typedef __attribute__((ext_vector_type(16))) __bf16 v16bf;
typedef __attribute__((ext_vector_type(8)))  __bf16 v8bf;
typedef __attribute__((ext_vector_type(8)))  float  v8f;

#define B_   32
#define S_   512
#define E_   300
#define F_   512
#define T_   512
#define FF_  1536
#define KC_  960            // 3*E = 900 padded up to a multiple of 64
#define M_   16384          // B*S == B*T
#define LBL_ 8
#define LDT  72             // padded LDS row stride in bf16 elements (144 B)

static __device__ __forceinline__ unsigned short f2bf_u(float x) {
  unsigned u = __builtin_bit_cast(unsigned, x);
  u += 0x7FFFu + ((u >> 16) & 1u);          // round-to-nearest-even
  return (unsigned short)(u >> 16);
}
static __device__ __forceinline__ float bfu2f(unsigned short s) {
  unsigned u = ((unsigned)s) << 16;
  return __builtin_bit_cast(float, u);
}

// 4 x b128 async global->LDS: same INST_OFFSET applies to both LDS dest and
// global source, so one LDS-offset VGPR + one 64-bit address pair covers 64 B.
static __device__ __forceinline__ void async_copy64B(unsigned lds_off,
                                                     unsigned long long gaddr) {
  asm volatile(
      "global_load_async_to_lds_b128 %0, %1, off\n\t"
      "global_load_async_to_lds_b128 %0, %1, off offset:16\n\t"
      "global_load_async_to_lds_b128 %0, %1, off offset:32\n\t"
      "global_load_async_to_lds_b128 %0, %1, off offset:48"
      :: "v"(lds_off), "v"(gaddr) : "memory");
}
static __device__ __forceinline__ void wait_async0() {
  asm volatile("s_wait_asynccnt 0x0" ::: "memory");
}
static __device__ __forceinline__ unsigned lds_off_u32(const void* p) {
  // flat LDS addresses carry the wave-relative LDS byte offset in addr[31:0]
  return (unsigned)(unsigned long long)p;
}

// ---- weight repack: conv_w [F,1,3,E] f32 -> [F, KC] bf16 (zero padded) -----
__global__ void prep_conv_w_kernel(const float* __restrict__ cw,
                                   unsigned short* __restrict__ Wc) {
  int i = blockIdx.x * 256 + threadIdx.x;
  if (i >= F_ * KC_) return;
  int f = i / KC_, c = i % KC_;
  float v = 0.f;
  if (c < 3 * E_) v = cw[f * (3 * E_) + c];   // [f, 0, k, e] is f*900 + c
  Wc[i] = f2bf_u(v);
}

// ---- weight repack: fc1_w [FF,FF] f32 -> bf16 ------------------------------
__global__ void prep_fc1_w_kernel(const float* __restrict__ w,
                                  unsigned short* __restrict__ wb) {
  int i = blockIdx.x * 256 + threadIdx.x;     // FF*FF is exact multiple of 256
  wb[i] = f2bf_u(w[i]);
}

// ---- embedding gather + im2col -> X2 [B*S, KC] bf16 ------------------------
__global__ void im2col_embed_kernel(const int* __restrict__ word_batch,
                                    const float* __restrict__ emb,
                                    unsigned short* __restrict__ X2) {
  long long i = (long long)blockIdx.x * 256 + threadIdx.x;
  if (i >= (long long)M_ * KC_) return;
  int row = (int)(i / KC_), c = (int)(i % KC_);
  int b = row / S_, s = row % S_;
  float v = 0.f;
  if (c < 3 * E_) {
    int k = c / E_, e = c % E_;
    int s2 = s + k - 1;                       // padding = K//2 = 1
    if (s2 >= 0 && s2 < S_) {
      int w = word_batch[b * S_ + s2];
      v = emb[(long long)w * E_ + e];
    }
  }
  X2[i] = f2bf_u(v);
}

// ---- WMMA GEMM: C[M,N] = relu(A[M,KD] @ B[N,KD]^T + bias[N]) ---------------
// 128x128 block tile, BK=64, 8 waves each computing 2x4 16x16 WMMA tiles.
// Double-buffered LDS; next tile fetched with async global->LDS loads while
// the current tile feeds 16 WMMAs. EPI==0 -> f32 out, EPI==1 -> bf16 out.
template <int KD, int N, int EPI>
__global__ void __launch_bounds__(256)
gemm_bf16_wmma_kernel(const unsigned short* __restrict__ A,
                      const unsigned short* __restrict__ Bm,
                      const float* __restrict__ bias,
                      void* __restrict__ Cout) {
  constexpr int KT = KD / 64;
  __shared__ __align__(16) unsigned short ldsA[2][128 * LDT];
  __shared__ __align__(16) unsigned short ldsB[2][128 * LDT];

  const int tid  = threadIdx.x;
  const int lane = tid & 31;
  const int wid  = tid >> 5;
  const int wRow = wid >> 1;                  // 0..3  -> 32-row strip
  const int wCol = wid & 1;                   // 0..1  -> 64-col strip
  const int half = lane >> 4;                 // WMMA lane-half
  const int lrow = lane & 15;
  const int k0off = half * 8;                 // K 0..7  / 8..15 within substep
  const int k1off = 16 + half * 8;            // K 16..23 / 24..31
  const int mBase = blockIdx.y * 128;
  const int nBase = blockIdx.x * 128;

  // cooperative copy: each thread owns 64 B (32 bf16) per matrix per tile
  const int crow = tid >> 1;                  // 0..127
  const int ccol = (tid & 1) * 32;            // 0 or 32
  const unsigned short* gA = A + (size_t)(mBase + crow) * KD + ccol;
  const unsigned short* gB = Bm + (size_t)(nBase + crow) * KD + ccol;
  const unsigned ldsAoff[2] = {lds_off_u32(&ldsA[0][crow * LDT + ccol]),
                               lds_off_u32(&ldsA[1][crow * LDT + ccol])};
  const unsigned ldsBoff[2] = {lds_off_u32(&ldsB[0][crow * LDT + ccol]),
                               lds_off_u32(&ldsB[1][crow * LDT + ccol])};

  // prologue: kick off tile 0 into buffer 0
  async_copy64B(ldsAoff[0], (unsigned long long)gA);
  async_copy64B(ldsBoff[0], (unsigned long long)gB);

  v8f acc[2][4] = {};

  for (int kt = 0; kt < KT; ++kt) {
    wait_async0();          // this wave's LDS writes for tile kt are done
    __syncthreads();        // ... and every other wave's too
    if (kt + 1 < KT) {      // prefetch tile kt+1 into the other buffer
      int nb = (kt + 1) & 1;
      async_copy64B(ldsAoff[nb], (unsigned long long)(gA + (kt + 1) * 64));
      async_copy64B(ldsBoff[nb], (unsigned long long)(gB + (kt + 1) * 64));
    }
    const unsigned short* bufA = ldsA[kt & 1];
    const unsigned short* bufB = ldsB[kt & 1];
#pragma unroll
    for (int ks = 0; ks < 2; ++ks) {          // two K=32 substeps per tile
      v16bf fa[2], fb[4];
#pragma unroll
      for (int mi = 0; mi < 2; ++mi) {
        int rb = (wRow * 32 + mi * 16 + lrow) * LDT + ks * 32;
        v8bf lo = *reinterpret_cast<const v8bf*>(&bufA[rb + k0off]);
        v8bf hi = *reinterpret_cast<const v8bf*>(&bufA[rb + k1off]);
        fa[mi] = __builtin_shufflevector(lo, hi, 0, 1, 2, 3, 4, 5, 6, 7,
                                         8, 9, 10, 11, 12, 13, 14, 15);
      }
#pragma unroll
      for (int ni = 0; ni < 4; ++ni) {
        int rb = (wCol * 64 + ni * 16 + lrow) * LDT + ks * 32;
        v8bf lo = *reinterpret_cast<const v8bf*>(&bufB[rb + k0off]);
        v8bf hi = *reinterpret_cast<const v8bf*>(&bufB[rb + k1off]);
        fb[ni] = __builtin_shufflevector(lo, hi, 0, 1, 2, 3, 4, 5, 6, 7,
                                         8, 9, 10, 11, 12, 13, 14, 15);
      }
#pragma unroll
      for (int mi = 0; mi < 2; ++mi)
#pragma unroll
        for (int ni = 0; ni < 4; ++ni)
          acc[mi][ni] = __builtin_amdgcn_wmma_f32_16x16x32_bf16(
              false, fa[mi], false, fb[ni], (short)0, acc[mi][ni], false, false);
    }
  }

  // epilogue: bias + relu; C/D layout: VGPR r holds M = r + 8*half, N = lrow.
  float* Cf = (float*)Cout;
  unsigned short* Ch = (unsigned short*)Cout;
#pragma unroll
  for (int mi = 0; mi < 2; ++mi)
#pragma unroll
    for (int ni = 0; ni < 4; ++ni) {
      int col = nBase + wCol * 64 + ni * 16 + lrow;
      float bv = bias[col];
#pragma unroll
      for (int r = 0; r < 8; ++r) {
        int rowg = mBase + wRow * 32 + mi * 16 + half * 8 + r;
        float v = acc[mi][ni][r] + bv;
        v = v > 0.f ? v : 0.f;
        if (EPI == 0)
          Cf[(size_t)rowg * N + col] = v;
        else
          Ch[(size_t)rowg * N + col] = f2bf_u(v);
      }
    }
}

// ---- inclusive cumsum over S, feat [B,S,F] f32 -> cs [B,S+1,F] f32 ---------
__global__ void cumsum_kernel(const float* __restrict__ feat,
                              float* __restrict__ cs) {
  int b = blockIdx.y;
  int f = blockIdx.x * 256 + threadIdx.x;     // < F_
  float run = 0.f;
  cs[((size_t)b * (S_ + 1)) * F_ + f] = 0.f;
  for (int s = 0; s < S_; ++s) {
    run += feat[((size_t)b * S_ + s) * F_ + f];
    cs[((size_t)b * (S_ + 1) + s + 1) * F_ + f] = run;
  }
}

// ---- TOI pooling -> pooled [B*T, 3F] bf16 (order: feat_s | avg | feat_e) ---
// tois declared jnp.int64 in the reference -> read as 64-bit.
__global__ void pool_kernel(const float* __restrict__ feat,
                            const float* __restrict__ cs,
                            const long long* __restrict__ tois,
                            unsigned short* __restrict__ pooled) {
  int row = blockIdx.x;                       // b*T + t
  int b = row / T_;
  int start = (int)tois[(size_t)row * 2 + 0];
  int end   = (int)tois[(size_t)row * 2 + 1];
  float inv = 1.f / (float)(end - start);
  for (int f = threadIdx.x; f < F_; f += 256) {
    float fs = feat[((size_t)b * S_ + start) * F_ + f];
    float fe = feat[((size_t)b * S_ + end - 1) * F_ + f];
    float av = (cs[((size_t)b * (S_ + 1) + end) * F_ + f] -
                cs[((size_t)b * (S_ + 1) + start) * F_ + f]) * inv;
    size_t o = (size_t)row * FF_;
    pooled[o + f]          = f2bf_u(fs);
    pooled[o + F_ + f]     = f2bf_u(av);
    pooled[o + 2 * F_ + f] = f2bf_u(fe);
  }
}

// ---- FC2: logits[row, l] = h[row,:] . fc2_w[l,:] + b2[l] -------------------
__global__ void fc2_kernel(const unsigned short* __restrict__ h,
                           const float* __restrict__ w2,
                           const float* __restrict__ b2,
                           float* __restrict__ out) {
  __shared__ float hrow[FF_];
  int row = blockIdx.x;
  for (int i = threadIdx.x; i < FF_; i += 256)
    hrow[i] = bfu2f(h[(size_t)row * FF_ + i]);
  __syncthreads();
  int label = threadIdx.x >> 5;               // 8 waves -> 8 labels
  int lane  = threadIdx.x & 31;
  float sum = 0.f;
  for (int k = lane; k < FF_; k += 32)
    sum += hrow[k] * w2[label * FF_ + k];
  for (int off = 16; off > 0; off >>= 1)
    sum += __shfl_xor(sum, off, 32);
  if (lane == 0)
    out[(size_t)row * LBL_ + label] = sum + b2[label];
}

// ---- toi_section = cumsum([T]*B) -------------------------------------------
__global__ void sections_kernel(float* __restrict__ out) {
  int i = threadIdx.x;
  if (i < B_) out[(size_t)M_ * LBL_ + i] = (float)((i + 1) * T_);
}

extern "C" void kernel_launch(void* const* d_in, const int* in_sizes, int n_in,
                              void* d_out, int out_size, void* d_ws, size_t ws_size,
                              hipStream_t stream) {
  (void)in_sizes; (void)n_in; (void)out_size;

  const int*       word_batch = (const int*)d_in[0];
  const long long* tois       = (const long long*)d_in[1];   // jnp.int64
  const float*     word_emb   = (const float*)d_in[2];
  const float*     conv_w     = (const float*)d_in[3];
  const float*     conv_b     = (const float*)d_in[4];
  const float*     fc1_w      = (const float*)d_in[5];
  const float*     fc1_b      = (const float*)d_in[6];
  const float*     fc2_w      = (const float*)d_in[7];
  const float*     fc2_b      = (const float*)d_in[8];
  float*           out        = (float*)d_out;
  char*            ws         = (char*)d_ws;

  // workspace layout (bytes), with lifetime-safe overlays:
  //  Wc    : conv weights bf16 [512, 960]
  //  fc1wb : fc1 weights bf16 [1536, 1536]
  //  feat  : conv output f32 [B,S,F]      (overlaid later by h)
  //  cs    : cumsum f32 [B,S+1,F]         (overlaid later by h)
  //  X2    : im2col bf16 [B*S, 960]       (overlaid later by pooled)
  const size_t off_Wc     = 0;
  const size_t off_fc1    = off_Wc  + (size_t)F_ * KC_ * 2;          //   983040
  const size_t off_feat   = off_fc1 + (size_t)FF_ * FF_ * 2;         //  5701632
  const size_t off_cs     = off_feat + (size_t)M_ * F_ * 4;          // 39256064
  const size_t off_X2     = off_cs + (size_t)B_ * (S_ + 1) * F_ * 4; // 72876032
  const size_t off_pooled = off_X2;     // X2 dead after conv GEMM
  const size_t off_h      = off_feat;   // feat+cs dead after pooling
  const size_t need = off_X2 + (size_t)M_ * FF_ * 2;                 // ~118 MB
  if (ws_size < need) return;

  unsigned short* Wc     = (unsigned short*)(ws + off_Wc);
  unsigned short* fc1wb  = (unsigned short*)(ws + off_fc1);
  float*          feat   = (float*)(ws + off_feat);
  float*          cs     = (float*)(ws + off_cs);
  unsigned short* X2     = (unsigned short*)(ws + off_X2);
  unsigned short* pooled = (unsigned short*)(ws + off_pooled);
  unsigned short* h      = (unsigned short*)(ws + off_h);

  // 1) weight repacks
  prep_conv_w_kernel<<<(F_ * KC_) / 256, 256, 0, stream>>>(conv_w, Wc);
  prep_fc1_w_kernel<<<(FF_ * FF_) / 256, 256, 0, stream>>>(fc1_w, fc1wb);

  // 2) embedding gather + im2col
  im2col_embed_kernel<<<(M_ * KC_) / 256, 256, 0, stream>>>(word_batch, word_emb, X2);

  // 3) conv as GEMM: feat[B*S, F] = relu(X2 @ Wc^T + conv_b)
  gemm_bf16_wmma_kernel<KC_, F_, 0><<<dim3(F_ / 128, M_ / 128), 256, 0, stream>>>(
      X2, Wc, conv_b, (void*)feat);

  // 4) cumsum over sequence
  cumsum_kernel<<<dim3(F_ / 256, B_), 256, 0, stream>>>(feat, cs);

  // 5) TOI pooling -> pooled bf16 [B*T, 3F]
  pool_kernel<<<M_, 256, 0, stream>>>(feat, cs, tois, pooled);

  // 6) FC1: h[B*T, FF] = relu(pooled @ fc1_w^T + fc1_b), bf16 out
  gemm_bf16_wmma_kernel<FF_, FF_, 1><<<dim3(FF_ / 128, M_ / 128), 256, 0, stream>>>(
      pooled, fc1wb, fc1_b, (void*)h);

  // 7) FC2 -> logits f32 [B*T, 8]
  fc2_kernel<<<M_, 256, 0, stream>>>(h, fc2_w, fc2_b, out);

  // 8) toi_section tail
  sections_kernel<<<1, 32, 0, stream>>>(out);
}
